// ConvLayer_31009663877367
// MI455X (gfx1250) — matrix operand order repaired and checked by reading, not compiled, not run on hardware
//
#include <hip/hip_runtime.h>
#include <hip/hip_bf16.h>
#include <math.h>

typedef __attribute__((ext_vector_type(16))) _Float16 v16h;
typedef __attribute__((ext_vector_type(8)))  _Float16 v8h;
typedef __attribute__((ext_vector_type(8)))  float    v8f;

#define NN 50000
#define NE 400000

#define INV_SQ3    0.5773502691896258f
#define INV_SQ2    0.7071067811865476f
#define INV_SQRT20 0.22360679774997896f   // 1/sqrt(16+4)
#define INV_SQRT24 0.20412414523193154f   // 1/sqrt(16+8)
#define INV_SQRT68 0.12126781251816650f   // 1/sqrt(8*8+2*2)
#define INV_SQRT32 0.17677669529663687f
#define INV_SQRT8  0.35355339059327373f
#define EPSN       1e-5f

// ---------------- WMMA helpers ----------------
__device__ __forceinline__ v8f wmma16(v16h a, v16h b, v8f c) {
  return __builtin_amdgcn_wmma_f32_16x16x32_f16(false, a, false, b, (short)0, c, false, false);
}
__device__ __forceinline__ v8f zf8() {
  v8f z;
#pragma unroll
  for (int i = 0; i < 8; ++i) z[i] = 0.f;
  return z;
}
__device__ __forceinline__ float gelu_t(float x) {
  float x3 = x * x * x;
  return 0.5f * x * (1.f + tanhf(0.7978845608028654f * (x + 0.044715f * x3)));
}
__device__ __forceinline__ void dswait() {
  asm volatile("s_wait_dscnt 0" ::: "memory");
}

// A fragment (16x32 f16): rows [row0, row0+16) of LDS row-major buffer, k-block kblk.
__device__ __forceinline__ v16h ldsA(const _Float16* buf, int row0, int lane, int kblk, int ldk) {
  int row = row0 + (lane & 15), half = (lane >> 4) & 1;
  int kb = kblk * 32 + half * 8;
  v8h lo = *(const v8h*)(buf + row * ldk + kb);
  v8h hi = *(const v8h*)(buf + row * ldk + kb + 16);
  v16h a;
#pragma unroll
  for (int j = 0; j < 8; ++j) { a[j] = lo[j]; a[8 + j] = hi[j]; }
  return a;
}
// B fragment (32x16 f16) from N-major f16 weights Wt[n][k] (Kin = row stride).
__device__ __forceinline__ v16h gB(const _Float16* Wt, int Kin, int lane, int kblk, int nblk) {
  int n = nblk * 16 + (lane & 15), half = (lane >> 4) & 1;
  const _Float16* p = Wt + (size_t)n * Kin + kblk * 32 + half * 8;
  v8h lo = *(const v8h*)p;
  v8h hi = *(const v8h*)(p + 16);
  v16h b;
#pragma unroll
  for (int j = 0; j < 8; ++j) { b[j] = lo[j]; b[8 + j] = hi[j]; }
  return b;
}

// 3-layer MLP for a 32-edge tile (2 M-tiles): 128 ->gelu 64 ->gelu 64 -> n3blks*16
// xbuf: 32x128 f16 inputs; hbuf: 32x64 f16 staging; obuf: 32x416 f16 outputs.
__device__ __forceinline__ void mlp3_32edges(
    int lane, const _Float16* xbuf,
    const _Float16* Wt1, const float* b1,
    const _Float16* Wt2, const float* b2,
    const _Float16* Wt3, const float* b3,
    int n3blks, _Float16* hbuf, _Float16* obuf) {
  int lcol = lane & 15, half = (lane >> 4) & 1;
  // ---- layer 1: K=128, M=32 ----
  v8f c[4][2];
#pragma unroll
  for (int nb = 0; nb < 4; ++nb) { c[nb][0] = zf8(); c[nb][1] = zf8(); }
#pragma unroll
  for (int kb = 0; kb < 4; ++kb) {
    v16h a0 = ldsA(xbuf, 0, lane, kb, 128);
    v16h a1 = ldsA(xbuf, 16, lane, kb, 128);
#pragma unroll
    for (int nb = 0; nb < 4; ++nb) {
      v16h b = gB(Wt1, 128, lane, kb, nb);
      c[nb][0] = wmma16(a0, b, c[nb][0]);
      c[nb][1] = wmma16(a1, b, c[nb][1]);
    }
  }
#pragma unroll
  for (int nb = 0; nb < 4; ++nb) {
    float bia = b1[nb * 16 + lcol];
#pragma unroll
    for (int mt = 0; mt < 2; ++mt)
#pragma unroll
      for (int r = 0; r < 8; ++r) {
        int row = mt * 16 + (half ? (r + 8) : r);
        hbuf[row * 64 + nb * 16 + lcol] = (_Float16)gelu_t(c[nb][mt][r] + bia);
      }
  }
  dswait();
  // ---- layer 2: K=64 ----
  v16h a2[2][2];
#pragma unroll
  for (int kb = 0; kb < 2; ++kb) {
    a2[kb][0] = ldsA(hbuf, 0, lane, kb, 64);
    a2[kb][1] = ldsA(hbuf, 16, lane, kb, 64);
  }
#pragma unroll
  for (int nb = 0; nb < 4; ++nb) { c[nb][0] = zf8(); c[nb][1] = zf8(); }
#pragma unroll
  for (int kb = 0; kb < 2; ++kb)
#pragma unroll
    for (int nb = 0; nb < 4; ++nb) {
      v16h b = gB(Wt2, 64, lane, kb, nb);
      c[nb][0] = wmma16(a2[kb][0], b, c[nb][0]);
      c[nb][1] = wmma16(a2[kb][1], b, c[nb][1]);
    }
  dswait();
#pragma unroll
  for (int nb = 0; nb < 4; ++nb) {
    float bia = b2[nb * 16 + lcol];
#pragma unroll
    for (int mt = 0; mt < 2; ++mt)
#pragma unroll
      for (int r = 0; r < 8; ++r) {
        int row = mt * 16 + (half ? (r + 8) : r);
        hbuf[row * 64 + nb * 16 + lcol] = (_Float16)gelu_t(c[nb][mt][r] + bia);
      }
  }
  dswait();
  // ---- layer 3: K=64 -> n3blks*16 channels, f16 out ----
  v16h a3[2][2];
#pragma unroll
  for (int kb = 0; kb < 2; ++kb) {
    a3[kb][0] = ldsA(hbuf, 0, lane, kb, 64);
    a3[kb][1] = ldsA(hbuf, 16, lane, kb, 64);
  }
  for (int nb = 0; nb < n3blks; ++nb) {
    v8f e0 = zf8(), e1 = zf8();
#pragma unroll
    for (int kb = 0; kb < 2; ++kb) {
      v16h b = gB(Wt3, 64, lane, kb, nb);
      e0 = wmma16(a3[kb][0], b, e0);
      e1 = wmma16(a3[kb][1], b, e1);
    }
    float bia = b3[nb * 16 + lcol];
#pragma unroll
    for (int r = 0; r < 8; ++r) {
      int row = half ? (r + 8) : r;
      obuf[row * 416 + nb * 16 + lcol] = (_Float16)(e0[r] + bia);
      obuf[(row + 16) * 416 + nb * 16 + lcol] = (_Float16)(e1[r] + bia);
    }
  }
}

__device__ __forceinline__ float atomicMaxF32(float* addr, float val) {
  int* a = (int*)addr;
  int old = __hip_atomic_load(a, __ATOMIC_RELAXED, __HIP_MEMORY_SCOPE_AGENT);
  while (__int_as_float(old) < val) {
    int assumed = old;
    old = atomicCAS(a, assumed, __float_as_int(val));
    if (old == assumed) break;
  }
  return __int_as_float(old);
}

// ---------------- kernels ----------------

// transpose+convert: src (K x N f32, row-major) -> dst (N x K f16)
__global__ void k_wconv(const float* __restrict__ src, _Float16* __restrict__ dst,
                        int K, int N, int total) {
  for (int o = blockIdx.x * blockDim.x + threadIdx.x; o < total; o += gridDim.x * blockDim.x) {
    int n = o / K, k = o - n * K;
    dst[o] = (_Float16)src[k * N + n];
  }
}

// per-node transform + softmax-state init + stats zero
__global__ void k_nodes(const float* __restrict__ nd,
                        const float* __restrict__ Wi0, const float* __restrict__ Wi1,
                        const float* __restrict__ Wq0, const float* __restrict__ Wq1,
                        float* __restrict__ s_t, float* __restrict__ v_t,
                        float* __restrict__ q0, float* __restrict__ q1,
                        float* __restrict__ mbuf, float* __restrict__ zbuf,
                        float* __restrict__ agg, float* __restrict__ stats) {
  int n = blockIdx.x * blockDim.x + threadIdx.x;
  if (n < 144) stats[n] = 0.f;
  if (n >= NN) return;
  const float* x = nd + (size_t)n * 56;
  float st[16];
#pragma unroll
  for (int o = 0; o < 16; ++o) {
    float acc = 0.f;
#pragma unroll
    for (int i = 0; i < 32; ++i) acc += x[i] * Wi0[i * 16 + o];
    st[o] = acc * INV_SQRT32;
    s_t[(size_t)n * 16 + o] = st[o];
  }
  float vt[12];
#pragma unroll
  for (int o = 0; o < 4; ++o)
#pragma unroll
    for (int m = 0; m < 3; ++m) {
      float acc = 0.f;
#pragma unroll
      for (int i = 0; i < 8; ++i) acc += x[32 + i * 3 + m] * Wi1[i * 4 + o];
      vt[o * 3 + m] = acc * INV_SQRT8;
      v_t[(size_t)n * 12 + o * 3 + m] = vt[o * 3 + m];
    }
#pragma unroll
  for (int k = 0; k < 8; ++k) {
    float acc = 0.f;
#pragma unroll
    for (int o = 0; o < 16; ++o) acc += st[o] * Wq0[o * 8 + k];
    q0[(size_t)n * 8 + k] = acc * 0.25f;
  }
#pragma unroll
  for (int k = 0; k < 2; ++k)
#pragma unroll
    for (int m = 0; m < 3; ++m) {
      float acc = 0.f;
#pragma unroll
      for (int o = 0; o < 4; ++o) acc += vt[o * 3 + m] * Wq1[o * 2 + k];
      q1[(size_t)n * 6 + k * 3 + m] = acc * 0.5f;
    }
  mbuf[n] = -INFINITY;
  zbuf[n] = 0.f;
#pragma unroll
  for (int c = 0; c < 28; ++c) agg[(size_t)n * 28 + c] = 0.f;
}

// edge kernel: 32 edges per wave; WMMA MLPs + tp_fc + logits + messages
__global__ __launch_bounds__(32) void k_edges(
    const float* __restrict__ edge_data, const int* __restrict__ eidx,
    const float* __restrict__ sh,
    const float* __restrict__ s_t, const float* __restrict__ v_t,
    const float* __restrict__ q0, const float* __restrict__ q1,
    const _Float16* __restrict__ WtK1, const _Float16* __restrict__ WtK2,
    const _Float16* __restrict__ WtK3, const _Float16* __restrict__ WtV1,
    const _Float16* __restrict__ WtV2, const _Float16* __restrict__ WtV3,
    const float* __restrict__ fck_b1, const float* __restrict__ fck_b2,
    const float* __restrict__ fck_b3, const float* __restrict__ fcv_b1,
    const float* __restrict__ fcv_b2, const float* __restrict__ fcv_b3,
    const float* __restrict__ Wd00, const float* __restrict__ Wd11,
    float* __restrict__ mbuf, float* __restrict__ logits, float* __restrict__ msg) {
  __shared__ _Float16 xbuf[32 * 128];  // edge features, f16
  __shared__ _Float16 hbuf[32 * 64];   // hidden staging
  __shared__ _Float16 wkv[32 * 416];   // layer-3 outputs (wk then wv)
  int lane = threadIdx.x;
  int e0 = blockIdx.x * 32;

  // stage 32x128 edge features (f32 -> f16) into LDS
  for (int cch = lane; cch < 512; cch += 32) {
    int row = cch >> 4;
    int col = (cch & 15) << 3;
    const float4* p = (const float4*)(edge_data + (size_t)(e0 + row) * 128 + col);
    float4 f0 = p[0], f1 = p[1];
    v8h h;
    h[0] = (_Float16)f0.x; h[1] = (_Float16)f0.y; h[2] = (_Float16)f0.z; h[3] = (_Float16)f0.w;
    h[4] = (_Float16)f1.x; h[5] = (_Float16)f1.y; h[6] = (_Float16)f1.z; h[7] = (_Float16)f1.w;
    *(v8h*)(&xbuf[row * 128 + col]) = h;
  }
  dswait();

  // ---- fck MLP ----
  mlp3_32edges(lane, xbuf, WtK1, fck_b1, WtK2, fck_b2, WtK3, fck_b3, 13, hbuf, wkv);
  dswait();

  // per-lane edge quantities (one edge per lane, kept live across both phases)
  int e = e0 + lane;
  int src = eidx[e];
  int dst = eidx[NE + e];
  float s[16];
#pragma unroll
  for (int i = 0; i < 16; ++i) s[i] = s_t[(size_t)src * 16 + i];
  float v[4][3];
#pragma unroll
  for (int i = 0; i < 4; ++i)
#pragma unroll
    for (int m = 0; m < 3; ++m) v[i][m] = v_t[(size_t)src * 12 + i * 3 + m];
  float sh0 = sh[(size_t)e * 4 + 0];
  float s1[3] = {sh[(size_t)e * 4 + 1], sh[(size_t)e * 4 + 2], sh[(size_t)e * 4 + 3]};
  float sv1[4], cr[4][3];
#pragma unroll
  for (int i = 0; i < 4; ++i) {
    sv1[i] = v[i][0] * s1[0] + v[i][1] * s1[1] + v[i][2] * s1[2];
    cr[i][0] = (v[i][1] * s1[2] - v[i][2] * s1[1]) * INV_SQ2;
    cr[i][1] = (v[i][2] * s1[0] - v[i][0] * s1[2]) * INV_SQ2;
    cr[i][2] = (v[i][0] * s1[1] - v[i][1] * s1[0]) * INV_SQ2;
  }

  // ---- tp_fc phase 1: k0/k1 + attention logit (reads wkv cols 0..207) ----
  {
    const _Float16* wk = &wkv[lane * 416];
    float k0[8];
#pragma unroll
    for (int o = 0; o < 8; ++o) {
      float a0 = 0.f, a1s = 0.f;
#pragma unroll
      for (int i = 0; i < 16; ++i) a0 += s[i] * (float)wk[i * 8 + o];
#pragma unroll
      for (int i = 0; i < 4; ++i) a1s += sv1[i] * (float)wk[128 + i * 8 + o];
      k0[o] = (sh0 * a0 + INV_SQ3 * a1s) * INV_SQRT20;
    }
    float k1[2][3];
#pragma unroll
    for (int o = 0; o < 2; ++o) {
      float sw = 0.f;
#pragma unroll
      for (int i = 0; i < 16; ++i) sw += s[i] * (float)wk[160 + i * 2 + o];
#pragma unroll
      for (int m = 0; m < 3; ++m) {
        float vw = 0.f, cw = 0.f;
#pragma unroll
        for (int i = 0; i < 4; ++i) {
          vw += v[i][m] * (float)wk[192 + i * 2 + o];
          cw += cr[i][m] * (float)wk[200 + i * 2 + o];
        }
        k1[o][m] = (s1[m] * sw + sh0 * vw + cw) * INV_SQRT24;
      }
    }
    float q0d[8];
#pragma unroll
    for (int k = 0; k < 8; ++k) q0d[k] = q0[(size_t)dst * 8 + k];
    float q1d[2][3];
#pragma unroll
    for (int i = 0; i < 2; ++i)
#pragma unroll
      for (int m = 0; m < 3; ++m) q1d[i][m] = q1[(size_t)dst * 6 + i * 3 + m];
    float dot0 = 0.f;
#pragma unroll
    for (int i = 0; i < 8; ++i) {
      float t = 0.f;
#pragma unroll
      for (int j = 0; j < 8; ++j) t += Wd00[i * 8 + j] * k0[j];
      dot0 += q0d[i] * t;
    }
    float dot1 = 0.f;
#pragma unroll
    for (int i = 0; i < 2; ++i)
#pragma unroll
      for (int j = 0; j < 2; ++j) {
        float t = q1d[i][0] * k1[j][0] + q1d[i][1] * k1[j][1] + q1d[i][2] * k1[j][2];
        dot1 += Wd11[i * 2 + j] * t;
      }
    float ae = (dot0 + dot1 * INV_SQ3) * INV_SQRT68;
    logits[e] = ae;
    atomicMaxF32(&mbuf[dst], ae);
  }

  // ---- fcv MLP (reuses wkv buffer, cols 0..415) ----
  mlp3_32edges(lane, xbuf, WtV1, fcv_b1, WtV2, fcv_b2, WtV3, fcv_b3, 26, hbuf, wkv);
  dswait();

  // ---- tp_fc phase 2: vv0/vv1 -> messages ----
  {
    const _Float16* wv = &wkv[lane * 416];
    float vv0[16];
#pragma unroll
    for (int o = 0; o < 16; ++o) {
      float a0 = 0.f, a1s = 0.f;
#pragma unroll
      for (int i = 0; i < 16; ++i) a0 += s[i] * (float)wv[i * 16 + o];
#pragma unroll
      for (int i = 0; i < 4; ++i) a1s += sv1[i] * (float)wv[256 + i * 16 + o];
      vv0[o] = (sh0 * a0 + INV_SQ3 * a1s) * INV_SQRT20;
    }
    float vv1[4][3];
#pragma unroll
    for (int o = 0; o < 4; ++o) {
      float sw = 0.f;
#pragma unroll
      for (int i = 0; i < 16; ++i) sw += s[i] * (float)wv[320 + i * 4 + o];
#pragma unroll
      for (int m = 0; m < 3; ++m) {
        float vw = 0.f, cw = 0.f;
#pragma unroll
        for (int i = 0; i < 4; ++i) {
          vw += v[i][m] * (float)wv[384 + i * 4 + o];
          cw += cr[i][m] * (float)wv[400 + i * 4 + o];
        }
        vv1[o][m] = (s1[m] * sw + sh0 * vw + cw) * INV_SQRT24;
      }
    }
    float* mrow = msg + (size_t)e * 28;
#pragma unroll
    for (int o = 0; o < 16; ++o) mrow[o] = vv0[o];
#pragma unroll
    for (int o = 0; o < 4; ++o)
#pragma unroll
      for (int m = 0; m < 3; ++m) mrow[16 + o * 3 + m] = vv1[o][m];
  }
}

__global__ void k_expsum(const int* __restrict__ eidx, float* __restrict__ logits,
                         const float* __restrict__ mbuf, float* __restrict__ zbuf) {
  int e = blockIdx.x * blockDim.x + threadIdx.x;
  if (e >= NE) return;
  int dst = eidx[NE + e];
  float p = __expf(logits[e] - mbuf[dst]);
  logits[e] = p;
  atomicAdd(&zbuf[dst], p);
}

__global__ void k_scatter(const int* __restrict__ eidx, const float* __restrict__ logits,
                          const float* __restrict__ zbuf, const float* __restrict__ msg,
                          float* __restrict__ agg) {
  int e = blockIdx.x * blockDim.x + threadIdx.x;
  if (e >= NE) return;
  int dst = eidx[NE + e];
  float w = logits[e] / zbuf[dst];
  const float* mr = msg + (size_t)e * 28;
  float* ar = agg + (size_t)dst * 28;
#pragma unroll
  for (int c = 0; c < 28; ++c) atomicAdd(&ar[c], mr[c] * w);
}

__global__ void k_nodeout(const float* __restrict__ agg, const float* __restrict__ nd,
                          const float* __restrict__ Wo0, const float* __restrict__ Wo1,
                          float* __restrict__ outbuf, float* __restrict__ stats) {
  __shared__ float sred[144];
  int t = threadIdx.x;
  if (t < 144) sred[t] = 0.f;
  __syncthreads();
  int n = blockIdx.x * blockDim.x + t;
  if (n < NN) {
    const float* a0 = agg + (size_t)n * 28;
    const float* a1 = a0 + 16;
    const float* x = nd + (size_t)n * 56;
    float fs[64];
#pragma unroll
    for (int f = 0; f < 64; ++f) {
      float acc = 0.f;
#pragma unroll
      for (int o = 0; o < 16; ++o) acc += a0[o] * Wo0[o * 64 + f];
      fs[f] = acc * 0.25f;
      if (f < 56) fs[f] += x[f];
    }
    float fv[48];
#pragma unroll
    for (int f = 0; f < 16; ++f)
#pragma unroll
      for (int m = 0; m < 3; ++m) {
        float acc = 0.f;
#pragma unroll
        for (int o = 0; o < 4; ++o) acc += a1[o * 3 + m] * Wo1[o * 16 + f];
        fv[f * 3 + m] = acc * 0.5f;
      }
    float* orow = outbuf + (size_t)n * 112;
#pragma unroll
    for (int f = 0; f < 64; ++f) orow[f] = fs[f];
#pragma unroll
    for (int j = 0; j < 48; ++j) orow[64 + j] = fv[j];
#pragma unroll
    for (int f = 0; f < 64; ++f) {
      atomicAdd(&sred[f], fs[f]);
      atomicAdd(&sred[64 + f], fs[f] * fs[f]);
    }
#pragma unroll
    for (int j = 0; j < 16; ++j) {
      float nm = fv[3 * j] * fv[3 * j] + fv[3 * j + 1] * fv[3 * j + 1] +
                 fv[3 * j + 2] * fv[3 * j + 2];
      atomicAdd(&sred[128 + j], nm);
    }
  }
  __syncthreads();
  if (t < 144) atomicAdd(&stats[t], sred[t]);
}

__global__ void k_final(const float* __restrict__ outbuf, const float* __restrict__ stats,
                        const float* __restrict__ bn_ws, const float* __restrict__ bn_bs,
                        const float* __restrict__ bn_wv, float* __restrict__ out) {
  int n = blockIdx.x * blockDim.x + threadIdx.x;
  if (n >= NN) return;
  const float invN = 1.f / (float)NN;
  const float* orow = outbuf + (size_t)n * 112;
  float* drow = out + (size_t)n * 112;
#pragma unroll
  for (int f = 0; f < 64; ++f) {
    float mu = stats[f] * invN;
    float var = stats[64 + f] * invN - mu * mu;
    drow[f] = (orow[f] - mu) * rsqrtf(var + EPSN) * bn_ws[f] + bn_bs[f];
  }
#pragma unroll
  for (int j = 0; j < 16; ++j) {
    float sc = bn_wv[j] * rsqrtf(stats[128 + j] * invN + EPSN);
#pragma unroll
    for (int m = 0; m < 3; ++m) drow[64 + j * 3 + m] = orow[64 + j * 3 + m] * sc;
  }
}

// ---------------- host ----------------
extern "C" void kernel_launch(void* const* d_in, const int* in_sizes, int n_in,
                              void* d_out, int out_size, void* d_ws, size_t ws_size,
                              hipStream_t stream) {
  const float* node_data = (const float*)d_in[0];
  const float* edge_data = (const float*)d_in[1];
  const float* sh        = (const float*)d_in[2];
  const float* Wi0 = (const float*)d_in[3];
  const float* Wi1 = (const float*)d_in[4];
  const float* Wq0 = (const float*)d_in[5];
  const float* Wq1 = (const float*)d_in[6];
  const float* Wo0 = (const float*)d_in[7];
  const float* Wo1 = (const float*)d_in[8];
  const float* fck_w1 = (const float*)d_in[9];
  const float* fck_b1 = (const float*)d_in[10];
  const float* fck_w2 = (const float*)d_in[11];
  const float* fck_b2 = (const float*)d_in[12];
  const float* fck_w3 = (const float*)d_in[13];
  const float* fck_b3 = (const float*)d_in[14];
  const float* fcv_w1 = (const float*)d_in[15];
  const float* fcv_b1 = (const float*)d_in[16];
  const float* fcv_w2 = (const float*)d_in[17];
  const float* fcv_b2 = (const float*)d_in[18];
  const float* fcv_w3 = (const float*)d_in[19];
  const float* fcv_b3 = (const float*)d_in[20];
  const float* Wd00 = (const float*)d_in[21];
  const float* Wd11 = (const float*)d_in[22];
  const float* bn_ws = (const float*)d_in[23];
  const float* bn_bs = (const float*)d_in[24];
  const float* bn_wv = (const float*)d_in[25];
  const int* eidx = (const int*)d_in[26];

  char* ws = (char*)d_ws;
  _Float16* WT = (_Float16*)ws;
  _Float16* WtK1 = WT + 0;       // 64x128 = 8192
  _Float16* WtK2 = WT + 8192;    // 64x64  = 4096
  _Float16* WtK3 = WT + 12288;   // 208x64 = 13312
  _Float16* WtV1 = WT + 25600;   // 64x128 = 8192
  _Float16* WtV2 = WT + 33792;   // 64x64  = 4096
  _Float16* WtV3 = WT + 37888;   // 416x64 = 26624 -> total 64512 halves
  float* fb     = (float*)(ws + 64512 * 2);
  float* s_t    = fb + 0;         // 800000
  float* v_t    = fb + 800000;    // 600000
  float* q0     = fb + 1400000;   // 400000
  float* q1     = fb + 1800000;   // 300000
  float* mbuf   = fb + 2100000;   // 50000
  float* zbuf   = fb + 2150000;   // 50000
  float* agg    = fb + 2200000;   // 1400000
  float* logits = fb + 3600000;   // 400000
  float* msg    = fb + 4000000;   // 11200000
  float* outbuf = fb + 15200000;  // 5600000
  float* stats  = fb + 20800000;  // 144

  // weight convert+transpose (f32 KxN -> f16 NxK)
  k_wconv<<<32, 256, 0, stream>>>(fck_w1, WtK1, 128, 64, 8192);
  k_wconv<<<16, 256, 0, stream>>>(fck_w2, WtK2, 64, 64, 4096);
  k_wconv<<<52, 256, 0, stream>>>(fck_w3, WtK3, 64, 208, 13312);
  k_wconv<<<32, 256, 0, stream>>>(fcv_w1, WtV1, 128, 64, 8192);
  k_wconv<<<16, 256, 0, stream>>>(fcv_w2, WtV2, 64, 64, 4096);
  k_wconv<<<104, 256, 0, stream>>>(fcv_w3, WtV3, 64, 416, 26624);

  k_nodes<<<(NN + 255) / 256, 256, 0, stream>>>(node_data, Wi0, Wi1, Wq0, Wq1,
                                                s_t, v_t, q0, q1, mbuf, zbuf, agg, stats);

  k_edges<<<NE / 32, 32, 0, stream>>>(edge_data, eidx, sh, s_t, v_t, q0, q1,
                                      WtK1, WtK2, WtK3, WtV1, WtV2, WtV3,
                                      fck_b1, fck_b2, fck_b3, fcv_b1, fcv_b2, fcv_b3,
                                      Wd00, Wd11, mbuf, logits, msg);

  k_expsum<<<(NE + 255) / 256, 256, 0, stream>>>(eidx, logits, mbuf, zbuf);
  k_scatter<<<(NE + 255) / 256, 256, 0, stream>>>(eidx, logits, zbuf, msg, agg);
  k_nodeout<<<(NN + 255) / 256, 256, 0, stream>>>(agg, node_data, Wo0, Wo1, outbuf, stats);
  k_final<<<(NN + 255) / 256, 256, 0, stream>>>(outbuf, stats, bn_ws, bn_bs, bn_wv,
                                                (float*)d_out);
}
